// Dynamic_Actor_48009144434975
// MI455X (gfx1250) — compile-verified, matrix operand order checked by reading, not run on hardware
//
#include <hip/hip_runtime.h>
#include <hip/hip_bf16.h>
#include <math.h>

// ---- problem constants ----
#define BB   2048
#define TT   32
#define DIN  1024
#define DHH  64
#define AA   64
#define HH   512
#define KLS  576      // DH + H
#define ROWS (BB*TT)  // 65536

typedef __attribute__((ext_vector_type(4)))  _Float16 v4h;
typedef __attribute__((ext_vector_type(8)))  _Float16 v8h;
typedef __attribute__((ext_vector_type(16))) _Float16 v16h;
typedef __attribute__((ext_vector_type(8)))  float    v8f;

__device__ __forceinline__ float sigf(float x){ return 1.0f/(1.0f+expf(-x)); }

// Load one 16x32 f16 tile (A layout; B uses same pattern on N-major weights).
// Lane l: row r=l&15, group g=l>>4; halves K = 8g..8g+7 and 16+8g..16+8g+7.
__device__ __forceinline__ v16h ldtile(const _Float16* base, int ld, int lane){
  int r = lane & 15, g = lane >> 4;
  const _Float16* p = base + (size_t)r*ld + g*8;
  union { v16h v; v8h h[2]; } u;
  u.h[0] = *(const v8h*)(p);
  u.h[1] = *(const v8h*)(p + 16);
  return u.v;
}

__device__ __forceinline__ v8f wmma_f16(v16h a, v16h b, v8f c){
  return __builtin_amdgcn_wmma_f32_16x16x32_f16(false, a, false, b, (short)0, c, false, false);
}

// ---- transpose f32 [K,N] -> f16 [N,ldt] ----
__global__ void __launch_bounds__(256, 1)
k_transpose(const float* __restrict__ W, _Float16* __restrict__ Wt,
            int K, int N, int ldt){
  int idx = blockIdx.x*blockDim.x + threadIdx.x;
  if (idx >= K*N) return;
  int n = idx / K, k = idx - n*K;
  Wt[(size_t)n*ldt + k] = (_Float16)W[(size_t)k*N + n];
}

// ---- per-row mean/rstd over f32 ----
__global__ void __launch_bounds__(256, 1)
k_rowstats_f32(const float* __restrict__ X, float* __restrict__ mu,
               float* __restrict__ rs, int ncols){
  __shared__ float sA[256], sB[256];
  int row = blockIdx.x, tid = threadIdx.x;
  const float* p = X + (size_t)row*ncols;
  float a = 0.f, b = 0.f;
  for (int k = tid; k < ncols; k += 256){ float v = p[k]; a += v; b += v*v; }
  sA[tid] = a; sB[tid] = b; __syncthreads();
  for (int s = 128; s > 0; s >>= 1){
    if (tid < s){ sA[tid] += sA[tid+s]; sB[tid] += sB[tid+s]; }
    __syncthreads();
  }
  if (tid == 0){
    float m = sA[0]/(float)ncols;
    float v = sB[0]/(float)ncols - m*m;
    mu[row] = m; rs[row] = rsqrtf(v + 1e-12f);
  }
}

// ---- per-row mean/rstd over f16 (LN3: row = batch, ncols = T*2H) ----
__global__ void __launch_bounds__(256, 1)
k_rowstats_f16(const _Float16* __restrict__ X, float* __restrict__ mu,
               float* __restrict__ rs, int ncols){
  __shared__ float sA[256], sB[256];
  int row = blockIdx.x, tid = threadIdx.x;
  const _Float16* p = X + (size_t)row*ncols;
  float a = 0.f, b = 0.f;
  for (int k = tid; k < ncols; k += 256){ float v = (float)p[k]; a += v; b += v*v; }
  sA[tid] = a; sB[tid] = b; __syncthreads();
  for (int s = 128; s > 0; s >>= 1){
    if (tid < s){ sA[tid] += sA[tid+s]; sB[tid] += sB[tid+s]; }
    __syncthreads();
  }
  if (tid == 0){
    float m = sA[0]/(float)ncols;
    float v = sB[0]/(float)ncols - m*m;
    mu[row] = m; rs[row] = rsqrtf(v + 1e-12f);
  }
}

// ---- LN1 + dense(1024->64) + bias + LN2 + relu -> x16 [ROWS,64] f16 ----
__global__ void __launch_bounds__(128, 1)
k_dense(const float* __restrict__ ud,
        const float* __restrict__ mu1, const float* __restrict__ rs1,
        const float* __restrict__ g1, const float* __restrict__ bb1,
        const _Float16* __restrict__ Wt, const float* __restrict__ bd,
        const float* __restrict__ g2, const float* __restrict__ bb2,
        _Float16* __restrict__ x16){
  __shared__ __align__(16) float zbuf[16][64];
  __shared__ float mu2[16], rs2[16];
  int row0 = blockIdx.x * 16;
  int tid = threadIdx.x, lane = tid & 31, w = tid >> 5;
  int r = lane & 15, gq = lane >> 4;
  int row = row0 + r;
  float m = mu1[row], sc = rs1[row];
  v8f acc = {};
  for (int kb = 0; kb < DIN; kb += 32){
    v16h a;
    const float* s0 = ud + (size_t)row*DIN + kb + gq*8;
#pragma unroll
    for (int i = 0; i < 8; i++){ int k = kb + gq*8 + i;
      a[i] = (_Float16)((s0[i]-m)*sc*g1[k] + bb1[k]); }
#pragma unroll
    for (int i = 0; i < 8; i++){ int k = kb + 16 + gq*8 + i;
      a[8+i] = (_Float16)((s0[16+i]-m)*sc*g1[k] + bb1[k]); }
    v16h bm = ldtile(Wt + (size_t)(w*16)*DIN + kb, DIN, lane);
    acc = wmma_f16(a, bm, acc);
  }
  int col = w*16 + r;
  int Mb = (lane < 16) ? 0 : 8;
#pragma unroll
  for (int j = 0; j < 8; j++) zbuf[Mb+j][col] = acc[j] + bd[col];
  __syncthreads();
  if (tid < 16){
    float a = 0.f, b = 0.f;
    for (int k = 0; k < 64; k++){ float v = zbuf[tid][k]; a += v; b += v*v; }
    float mm = a/64.f;
    mu2[tid] = mm; rs2[tid] = rsqrtf(b/64.f - mm*mm + 1e-12f);
  }
  __syncthreads();
#pragma unroll
  for (int j = 0; j < 8; j++){
    int M = Mb + j;
    float v = (zbuf[M][col]-mu2[M])*rs2[M]*g2[col] + bb2[col];
    x16[(size_t)(row0+M)*64 + col] = (_Float16)fmaxf(v, 0.f);
  }
}

// ---- fused bidirectional persistent LSTM ----
// grid = 128: blockIdx>>6 = direction, (blockIdx&63)*32 = batch tile (M=32).
// 512 threads = 16 waves; wave w owns hidden cols [32w,32w+32):
//   2 N-tiles x 4 gates x 2 row-tiles = 16 f32 accumulators; one B-tile load
//   feeds two WMMAs (row-tiles), halving L2 weight traffic per FLOP.
// Cell state c lives in registers across all T steps; h in single LDS buffer.
#define HLD 520
__global__ void __launch_bounds__(512, 1)
k_lstm_bi(const _Float16* __restrict__ x16,
          const _Float16* __restrict__ WfT, const _Float16* __restrict__ WbT,
          const float* __restrict__ bfw, const float* __restrict__ bbw,
          const int* __restrict__ au, _Float16* __restrict__ hcat){
  __shared__ __align__(16) _Float16 hbuf[32*HLD];
  __shared__ __align__(16) _Float16 xbuf[32*72];
  __shared__ int lens[32], tix[32];
  int dir = blockIdx.x >> 6;
  int b0 = (blockIdx.x & 63) * 32;
  const _Float16* Wt  = dir ? WbT : WfT;
  const float*    bias = dir ? bbw : bfw;
  int dirOff = dir ? HH : 0;

  int tid = threadIdx.x, lane = tid & 31, w = tid >> 5;
  int r = lane & 15, Mb = (lane < 16) ? 0 : 8;

  for (int i = tid; i < 32*HLD; i += 512) hbuf[i] = (_Float16)0.f;
  if (tid < 32) lens[tid] = au[b0 + tid];
  __syncthreads();

  v8f cst[2][2];   // [q-tile][row-tile]
#pragma unroll
  for (int q = 0; q < 2; q++)
#pragma unroll
    for (int mm = 0; mm < 2; mm++) cst[q][mm] = {};

  for (int s = 0; s < TT; s++){
    if (tid < 32){ int L = lens[tid]; tix[tid] = dir ? ((s < L) ? (L-1-s) : s) : s; }
    __syncthreads();
    // stage x_t (per-row time index) into LDS: 32 rows x 64 halves
    { int rw = tid >> 4, q = tid & 15;
      const _Float16* src = x16 + ((size_t)(b0+rw)*TT + tix[rw])*64 + q*4;
      *(v4h*)&xbuf[rw*72 + q*4] = *(const v4h*)src;
    }
    __syncthreads();

    v8f acc[2][4][2];           // [q][gate][row-tile]
#pragma unroll
    for (int q = 0; q < 2; q++)
#pragma unroll
      for (int gte = 0; gte < 4; gte++)
#pragma unroll
        for (int mm = 0; mm < 2; mm++) acc[q][gte][mm] = {};

    for (int kb = 0; kb < KLS; kb += 32){
      v16h a0, a1;
      if (kb < 64){
        a0 = ldtile(xbuf + kb,            72, lane);
        a1 = ldtile(xbuf + 16*72 + kb,    72, lane);
      } else {
        a0 = ldtile(hbuf + (kb-64),          HLD, lane);
        a1 = ldtile(hbuf + 16*HLD + (kb-64), HLD, lane);
      }
#pragma unroll
      for (int q = 0; q < 2; q++){
        int hc = (w*2 + q)*16;                       // hidden-tile base column
#pragma unroll
        for (int gte = 0; gte < 4; gte++){
          v16h bm = ldtile(Wt + (size_t)(hc + gte*HH)*KLS + kb, KLS, lane);
          acc[q][gte][0] = wmma_f16(a0, bm, acc[q][gte][0]);
          acc[q][gte][1] = wmma_f16(a1, bm, acc[q][gte][1]);
        }
      }
    }
    __syncthreads();   // all reads of hbuf (h_{s-1}) done before rewrite

#pragma unroll
    for (int q = 0; q < 2; q++){
      int hc = (w*2 + q)*16;
      int ncol = hc + r;
      float Bi = bias[ncol], Bj = bias[HH+ncol];
      float Bf = bias[2*HH+ncol] + 1.0f, Bo = bias[3*HH+ncol];
#pragma unroll
      for (int mm = 0; mm < 2; mm++){
#pragma unroll
        for (int j = 0; j < 8; j++){
          float zi = acc[q][0][mm][j] + Bi;
          float zj = acc[q][1][mm][j] + Bj;
          float zf = acc[q][2][mm][j] + Bf;
          float zo = acc[q][3][mm][j] + Bo;
          float c = cst[q][mm][j]*sigf(zf) + sigf(zi)*tanhf(zj);
          cst[q][mm][j] = c;
          float h = tanhf(c)*sigf(zo);
          hbuf[(Mb + j + 16*mm)*HLD + ncol] = (_Float16)h;
        }
      }
    }
    __syncthreads();   // h_s fully in hbuf

    // masked, time-remapped store of h_s to hcat: 32 rows x 512 halves
    { int rw = tid >> 4, ch = tid & 15;
      int L = lens[rw], tout = tix[rw];
      _Float16* dst = hcat + (((size_t)(b0+rw)*TT + tout)*(2*HH)) + dirOff + ch*32;
      const _Float16* sp = &hbuf[rw*HLD + ch*32];
      if (s < L){
#pragma unroll
        for (int m2 = 0; m2 < 4; m2++) *(v8h*)(dst + m2*8) = *(const v8h*)(sp + m2*8);
      } else {
        v8h z = {};
#pragma unroll
        for (int m2 = 0; m2 < 4; m2++) *(v8h*)(dst + m2*8) = z;
      }
    }
    __syncthreads();   // stores done before next step rewrites tix/xbuf
  }
}

// ---- LN3 + relu + conv(1024->64) + bias + tanh -> out f32 ----
__global__ void __launch_bounds__(128, 1)
k_conv(const _Float16* __restrict__ hcat,
       const float* __restrict__ mu3, const float* __restrict__ rs3,
       const float* __restrict__ g3, const float* __restrict__ bb3,
       const _Float16* __restrict__ Wt, const float* __restrict__ bc,
       float* __restrict__ out){
  int row0 = blockIdx.x * 16;
  int tid = threadIdx.x, lane = tid & 31, w = tid >> 5;
  int r = lane & 15, gq = lane >> 4;
  int row = row0 + r;
  int bidx = row >> 5;                 // row / T, T == 32
  float m = mu3[bidx], sc = rs3[bidx];
  const _Float16* src = hcat + (size_t)row*(2*HH);
  v8f acc = {};
  for (int kb = 0; kb < 2*HH; kb += 32){
    v16h a;
#pragma unroll
    for (int i = 0; i < 8; i++){ int k = kb + gq*8 + i;
      float v = ((float)src[k]-m)*sc*g3[k] + bb3[k];
      a[i] = (_Float16)fmaxf(v, 0.f); }
#pragma unroll
    for (int i = 0; i < 8; i++){ int k = kb + 16 + gq*8 + i;
      float v = ((float)src[k]-m)*sc*g3[k] + bb3[k];
      a[8+i] = (_Float16)fmaxf(v, 0.f); }
    v16h bm = ldtile(Wt + (size_t)(w*16)*(2*HH) + kb, 2*HH, lane);
    acc = wmma_f16(a, bm, acc);
  }
  int col = w*16 + r, Mb = (lane < 16) ? 0 : 8;
#pragma unroll
  for (int j = 0; j < 8; j++)
    out[(size_t)(row0+Mb+j)*AA + col] = tanhf(acc[j] + bc[col]);
}

extern "C" void kernel_launch(void* const* d_in, const int* in_sizes, int n_in,
                              void* d_out, int out_size, void* d_ws, size_t ws_size,
                              hipStream_t stream) {
  const float* ud  = (const float*)d_in[0];
  const int*   au  = (const int*)  d_in[2];
  const float* g1  = (const float*)d_in[4];
  const float* bb1 = (const float*)d_in[5];
  const float* Wd  = (const float*)d_in[6];
  const float* bd  = (const float*)d_in[7];
  const float* g2  = (const float*)d_in[8];
  const float* bb2 = (const float*)d_in[9];
  const float* Wfw = (const float*)d_in[10];
  const float* bfw = (const float*)d_in[11];
  const float* Wbw = (const float*)d_in[12];
  const float* bbw = (const float*)d_in[13];
  const float* g3  = (const float*)d_in[14];
  const float* bb3 = (const float*)d_in[15];
  const float* Wc  = (const float*)d_in[16];
  const float* bc  = (const float*)d_in[17];
  float* out = (float*)d_out;

  char* ws = (char*)d_ws;
  size_t off = 0;
  auto take = [&](size_t n) -> void* {
    void* p = ws + off;
    off = (off + n + 255) & ~(size_t)255;
    return p;
  };
  float*     mu1  = (float*)    take((size_t)ROWS*4);
  float*     rs1  = (float*)    take((size_t)ROWS*4);
  _Float16*  x16  = (_Float16*) take((size_t)ROWS*DHH*2);
  _Float16*  WdT  = (_Float16*) take((size_t)DHH*DIN*2);
  _Float16*  WfT  = (_Float16*) take((size_t)4*HH*KLS*2);
  _Float16*  WbT  = (_Float16*) take((size_t)4*HH*KLS*2);
  _Float16*  WcT  = (_Float16*) take((size_t)AA*2*HH*2);
  _Float16*  hcat = (_Float16*) take((size_t)ROWS*2*HH*2);
  float*     mu3  = (float*)    take((size_t)BB*4);
  float*     rs3  = (float*)    take((size_t)BB*4);

  // weight transposes into N-major f16
  k_transpose<<<(DIN*DHH + 255)/256, 256, 0, stream>>>(Wd, WdT, DIN, DHH, DIN);
  k_transpose<<<(KLS*4*HH + 255)/256, 256, 0, stream>>>(Wfw, WfT, KLS, 4*HH, KLS);
  k_transpose<<<(KLS*4*HH + 255)/256, 256, 0, stream>>>(Wbw, WbT, KLS, 4*HH, KLS);
  k_transpose<<<(2*HH*AA + 255)/256, 256, 0, stream>>>(Wc, WcT, 2*HH, AA, 2*HH);

  // LN1 stats, then fused LN1+dense+LN2+relu (WMMA)
  k_rowstats_f32<<<ROWS, 256, 0, stream>>>(ud, mu1, rs1, DIN);
  k_dense<<<ROWS/16, 128, 0, stream>>>(ud, mu1, rs1, g1, bb1, WdT, bd, g2, bb2, x16);

  // fused bidirectional LSTM: both directions concurrent, M=32 per WG
  k_lstm_bi<<<128, 512, 0, stream>>>(x16, WfT, WbT, bfw, bbw, au, hcat);

  // LN3 stats per batch element, then fused LN3+relu+conv+tanh (WMMA)
  k_rowstats_f16<<<BB, 256, 0, stream>>>(hcat, mu3, rs3, TT*2*HH);
  k_conv<<<ROWS/16, 128, 0, stream>>>(hcat, mu3, rs3, g3, bb3, WcT, bc, out);
}
